// SimpleGATLayer_21775484191379
// MI455X (gfx1250) — matrix operand order repaired and checked by reading, not compiled, not run on hardware
//
#include <hip/hip_runtime.h>
#include <math.h>

typedef __attribute__((ext_vector_type(16))) _Float16 v16h;
typedef __attribute__((ext_vector_type(8)))  float    v8f;
typedef __attribute__((ext_vector_type(4)))  unsigned v4u;
typedef __attribute__((ext_vector_type(8)))  int      v8i;
typedef __attribute__((ext_vector_type(4)))  int      v4i;

static constexpr int BB = 8;      // batch
static constexpr int NN = 2048;   // nodes
static constexpr int FF = 128;    // features (in == out)
static constexpr float NEG_BIG = -9.0e15f;
static constexpr float SLOPE   = 0.01f;

#if __has_builtin(__builtin_amdgcn_tensor_load_to_lds) && __has_builtin(__builtin_amdgcn_s_wait_tensorcnt)
#define USE_TDM 1
#else
#define USE_TDM 0
#endif

__device__ __forceinline__ float leaky(float x) { return x >= 0.0f ? x : SLOPE * x; }

#if USE_TDM
// Issue a TDM 2D tile load: 64 rows x 128 halves of h (row stride 128 halves)
// into LDS. D# per CDNA5 ISA (group0/group1; groups 2/3
// zero for a 2D tensor). data_size=1 (2 bytes). type=2 ("image").
// This toolchain exposes the 6-arg builtin form:
//   (uint32x4 g0, int32x8 g1, int32x4 g2, int32x4 g3, int32x8 pad, i32 cpol)
__device__ __forceinline__ void tdm_load_h_tile(const _Float16* gsrc, _Float16* lds_dst) {
  unsigned long long ga = (unsigned long long)gsrc;
  unsigned lds_off = (unsigned)(unsigned long long)lds_dst;   // low 32b of generic = LDS byte offset
  v4u g0 = { 1u,                                   // count=1, user descriptor
             lds_off,                               // lds_addr [63:32]
             (unsigned)ga,                          // global_addr [95:64]
             (unsigned)((ga >> 32) & 0x01FFFFFFu) | 0x80000000u }; // addr[56:32] | type=2
  v8i g1 = { 0x00010000,   // data_size=2B; no multicast/pad/iterate
             0x00800000,   // tensor_dim0 = 128 (bits 63:48)
             0x40000000,   // tensor_dim1 = 16384 (bits 111:80; generous row count)
             0x00800000,   // tile_dim0 = 128 (bits 127:112)
             64,           // tile_dim1 = 64, tile_dim2 = 0
             128,          // tensor_dim0_stride = 128
             0, 0 };       // tensor_dim1_stride = 0 (unused for 2D)
  v4i z4 = {0, 0, 0, 0};
  v8i z8 = {0, 0, 0, 0, 0, 0, 0, 0};
  __builtin_amdgcn_tensor_load_to_lds(g0, g1, z4, z4, z8, 0);
}
#endif

// ---------------------------------------------------------------------------
// Kernel 1: h = feat @ W + b   (f32 inputs -> f16 h in workspace, f32 accum)
// ---------------------------------------------------------------------------
__global__ void __launch_bounds__(256)
gat_h_gemm(const float* __restrict__ feat, const float* __restrict__ W,
           const float* __restrict__ bias, _Float16* __restrict__ h16) {
  int bid  = blockIdx.x;
  int b    = bid >> 7;              // / (NN/16)
  int n0   = (bid & 127) << 4;
  int lane = threadIdx.x & 31;
  int wv   = threadIdx.x >> 5;
  int lo   = lane & 15, hi = lane >> 4;
  int f0   = wv * 16;

  v8f acc = {};
  for (int kc = 0; kc < FF / 32; ++kc) {
    int k0 = kc * 32;
    v16h a, bf;
    const float* fr = &feat[((size_t)b * NN + n0 + lo) * FF + k0];
#pragma unroll
    for (int j = 0; j < 8; ++j) {                 // A: 16x32 f16 layout (ISA 7.12.2)
      int base = (j < 4) ? (2 * j) : (16 + 2 * (j - 4));
      int k = base + hi * 8;
      a[2 * j]     = (_Float16)fr[k];
      a[2 * j + 1] = (_Float16)fr[k + 1];
    }
#pragma unroll
    for (int v = 0; v < 8; ++v) {                 // B: 32x16, rows striped across lanes
      int k = ((v < 4) ? (2 * v) : (16 + 2 * (v - 4))) + hi * 8;
      bf[2 * v]     = (_Float16)W[(size_t)(k0 + k) * FF + f0 + lo];
      bf[2 * v + 1] = (_Float16)W[(size_t)(k0 + k + 1) * FF + f0 + lo];
    }
    acc = __builtin_amdgcn_wmma_f32_16x16x32_f16(false, a, false, bf,
                                                 (short)0, acc, false, false);
  }

  int f = f0 + lo;
  float bb = bias[f];
#pragma unroll
  for (int i = 0; i < 8; ++i) {                   // C/D: row = i + hi*8, col = lo
    int row = i + hi * 8;
    h16[((size_t)b * NN + n0 + row) * FF + f] = (_Float16)(acc[i] + bb);
  }
}

// ---------------------------------------------------------------------------
// Kernel 2: e_src = h . a[:F], e_dst = h . a[F:]   (one thread per node)
// ---------------------------------------------------------------------------
__global__ void __launch_bounds__(256)
gat_edge_proj(const _Float16* __restrict__ h16, const float* __restrict__ avec,
              float* __restrict__ e_src, float* __restrict__ e_dst) {
  int node = blockIdx.x * 256 + threadIdx.x;
  if (node >= BB * NN) return;
  const _Float16* hr = &h16[(size_t)node * FF];
  float s = 0.0f, d = 0.0f;
#pragma unroll 4
  for (int f = 0; f < FF; ++f) {
    float v = (float)hr[f];
    s += v * avec[f];
    d += v * avec[FF + f];
  }
  e_src[node] = s;
  e_dst[node] = d;
}

// ---------------------------------------------------------------------------
// Kernel 3: fused score -> softmax -> (attn @ h) * mask.
// Block = 16 node rows of one batch. Pass 1: streaming online-softmax stats.
// Pass 2: 32 stages of 64 columns; h tile DMA'd into LDS by the Tensor Data
// Mover (double-buffered, overlapped with exp/adj work), exp-score tile in
// LDS f16, 2 WMMAs per wave per stage. 1/denom + mask folded at the end.
// ---------------------------------------------------------------------------
__global__ void __launch_bounds__(256)
gat_attn_agg(const int* __restrict__ adj, const float* __restrict__ mask,
             const _Float16* __restrict__ h16, const float* __restrict__ e_src,
             const float* __restrict__ e_dst, float* __restrict__ out) {
  __shared__ float    s_dst[NN];              // 8 KB: e_dst row for this batch
  __shared__ float    s_pm[16][16];
  __shared__ float    s_ps[16][16];
  __shared__ float    s_src[16];
  __shared__ float    s_rowmax[16];
  __shared__ float    s_invden[16];
  __shared__ _Float16 s_P[16 * 64];           // 2 KB: attention tile (A matrix)
  __shared__ _Float16 s_hbuf[2][64 * FF];     // 2 x 16 KB: double-buffered h tiles

  int bid  = blockIdx.x;
  int b    = bid >> 7;
  int n0   = (bid & 127) << 4;
  int t    = threadIdx.x;
  int lane = t & 31, wv = t >> 5;
  int lo   = lane & 15, hi = lane >> 4;

  for (int m = t; m < NN; m += 256) s_dst[m] = e_dst[(size_t)b * NN + m];
  if (t < 16) s_src[t] = e_src[(size_t)b * NN + n0 + t];
  __syncthreads();

  // ---- Pass 1: online softmax stats (16 threads per row, coalesced adj) ----
  {
    int r = t >> 4, c = t & 15;
    const int* ar = &adj[((size_t)b * NN + n0 + r) * NN];
    float srcv = s_src[r];
    float mrun = -INFINITY, srun = 0.0f;
    for (int k = 0; k < NN / 16; ++k) {
      int m   = c + (k << 4);
      float e = leaky(srcv + s_dst[m]);
      float sc = (ar[m] > 0) ? e : NEG_BIG;
      if (sc > mrun) { srun *= __expf(mrun - sc); mrun = sc; }
      srun += __expf(sc - mrun);
    }
    s_pm[r][c] = mrun;
    s_ps[r][c] = srun;
  }
  __syncthreads();
  if (t < 16) {
    float M = -INFINITY;
    for (int c = 0; c < 16; ++c) M = fmaxf(M, s_pm[t][c]);
    float D = 0.0f;
    for (int c = 0; c < 16; ++c) D += s_ps[t][c] * __expf(s_pm[t][c] - M);
    s_rowmax[t] = M;
    s_invden[t] = 1.0f / D;
  }
  __syncthreads();

  // ---- Pass 2: staged attention * h with WMMA ----
  v8f acc = {};
  int fb = wv * 16;
  constexpr int S = NN / 64;    // 32 stages

#if USE_TDM
  if (wv == 0) tdm_load_h_tile(&h16[(size_t)b * NN * FF], &s_hbuf[0][0]);
#endif

  for (int s = 0; s < S; ++s) {
    int m0   = s * 64;
    int bufi = s & 1;

#if USE_TDM
    // kick DMA for the next stage while this stage computes
    if (wv == 0 && s + 1 < S)
      tdm_load_h_tile(&h16[((size_t)b * NN + m0 + 64) * FF], &s_hbuf[(s + 1) & 1][0]);
#else
    {
      const unsigned int* hg = (const unsigned int*)&h16[((size_t)b * NN + m0) * FF];
      unsigned int* hl = (unsigned int*)&s_hbuf[bufi][0];
#pragma unroll
      for (int i = 0; i < 16; ++i) hl[t + 256 * i] = hg[t + 256 * i];
    }
#endif

    // prefetch next stage's adj tile (16 rows x 256B)
    if (s + 1 < S && t < 32) {
      int row = t >> 1, seg = t & 1;
      __builtin_prefetch(&adj[((size_t)b * NN + n0 + row) * NN + m0 + 64 + seg * 32], 0, 1);
    }

    // (a) 16x64 exp-score tile -> LDS (4 elems/thread, coalesced adj reads)
#pragma unroll
    for (int q = 0; q < 4; ++q) {
      int idx = q * 256 + t;
      int row = idx >> 6, col = idx & 63;
      int m   = m0 + col;
      int av  = adj[((size_t)b * NN + n0 + row) * NN + m];
      float sc = (av > 0) ? leaky(s_src[row] + s_dst[m]) : NEG_BIG;
      s_P[row * 64 + col] = (_Float16)__expf(sc - s_rowmax[row]);
    }

#if USE_TDM
    // oldest outstanding TDM (this stage's tile) must be done; the one just
    // issued for stage s+1 may remain in flight. Builtin needs a literal imm.
    if (s + 1 < S) __builtin_amdgcn_s_wait_tensorcnt(1);
    else           __builtin_amdgcn_s_wait_tensorcnt(0);
#endif
    __syncthreads();

    // (b) two 16x16x32 WMMAs per wave over this stage's 64 columns
    const _Float16* hb = &s_hbuf[bufi][0];
#pragma unroll
    for (int kcc = 0; kcc < 2; ++kcc) {
      int kb = kcc * 32;
      v16h a, bf;
#pragma unroll
      for (int j = 0; j < 8; ++j) {
        int base = (j < 4) ? (2 * j) : (16 + 2 * (j - 4));
        int k = kb + base + hi * 8;
        a[2 * j]     = s_P[lo * 64 + k];
        a[2 * j + 1] = s_P[lo * 64 + k + 1];
      }
#pragma unroll
      for (int v = 0; v < 8; ++v) {
        int k = kb + ((v < 4) ? (2 * v) : (16 + 2 * (v - 4))) + hi * 8;
        bf[2 * v]     = hb[k * FF + fb + lo];
        bf[2 * v + 1] = hb[(k + 1) * FF + fb + lo];
      }
      acc = __builtin_amdgcn_wmma_f32_16x16x32_f16(false, a, false, bf,
                                                   (short)0, acc, false, false);
    }
    __syncthreads();
  }

  // ---- epilogue: fold 1/denom and mask, store ----
  int f = fb + lo;
#pragma unroll
  for (int i = 0; i < 8; ++i) {
    int row  = i + hi * 8;
    int node = n0 + row;
    float val = acc[i] * s_invden[row] * mask[(size_t)b * NN + node];
    out[((size_t)b * NN + node) * FF + f] = val;
  }
}

// ---------------------------------------------------------------------------
extern "C" void kernel_launch(void* const* d_in, const int* in_sizes, int n_in,
                              void* d_out, int out_size, void* d_ws, size_t ws_size,
                              hipStream_t stream) {
  const float* feat = (const float*)d_in[0];   // (B,N,F) f32
  const int*   adj  = (const int*)d_in[1];     // (B,N,N) i32
  const float* mask = (const float*)d_in[2];   // (B,N)   f32
  const float* W    = (const float*)d_in[3];   // (F,F)   f32
  const float* bias = (const float*)d_in[4];   // (F,)    f32
  const float* avec = (const float*)d_in[5];   // (2F,)   f32
  float* out = (float*)d_out;

  // workspace layout: h16 (4 MB) | e_src (64 KB) | e_dst (64 KB)
  _Float16* h16   = (_Float16*)d_ws;
  float*    e_src = (float*)((char*)d_ws + (size_t)BB * NN * FF * sizeof(_Float16));
  float*    e_dst = e_src + (size_t)BB * NN;

  dim3 blk(256);
  gat_h_gemm   <<<dim3(BB * NN / 16),    blk, 0, stream>>>(feat, W, bias, h16);
  gat_edge_proj<<<dim3((BB * NN) / 256), blk, 0, stream>>>(h16, avec, e_src, e_dst);
  gat_attn_agg <<<dim3(BB * NN / 16),    blk, 0, stream>>>(adj, mask, h16, e_src, e_dst, out);
}
